// QuantLinear_15461882266230
// MI455X (gfx1250) — matrix operand order repaired and checked by reading, not compiled, not run on hardware
//
#include <hip/hip_runtime.h>

// ---------------------------------------------------------------------------
// QuantLinear (4-bit groupwise) + fused outlier linear + bias for MI455X
//
// y[M,N] = x[M,K] * W'[N,K]^T + bias,  M=8192, N=4096, K=4096
//   W'[n,k] = q[n,k]*scales[k/128,n] + scaled_zeros[k/128,n]
//             (+ oweight[n, k-3968] for k >= 3968  -- folded outlier term)
//
// Pass 1: convert x -> f16 (64 MiB) and dequantize W' -> f16 (32 MiB) in d_ws.
//         96 MiB working set fits in the 192 MB L2, so the GEMM streams tiles
//         from L2, not HBM.
// Pass 2: f16 WMMA GEMM, block tile 128x128, K-stage 64, double-buffered LDS
//         filled by GLOBAL_LOAD_ASYNC_TO_LDS_B128 (ASYNCcnt path, no VGPR
//         staging), 8 waves x (2x4 tiles of v_wmma_f32_16x16x32_f16).
// ---------------------------------------------------------------------------

typedef __attribute__((ext_vector_type(16))) _Float16 v16h;
typedef __attribute__((ext_vector_type(8)))  _Float16 v8h;
typedef __attribute__((ext_vector_type(8)))  float    v8f;
typedef __attribute__((ext_vector_type(4)))  float    v4f;

#define M_DIM 8192
#define N_DIM 4096
#define K_DIM 4096
#define QGROUP 128
#define NOUT 128

#define BM 128
#define BN 128
#define BK 64
#define LDK 72   // padded LDS row stride (f16 elems); 144B keeps 16B alignment

// ---------------- Pass 1a: x (fp32) -> xb (f16), 8 elems/thread -------------
__global__ __launch_bounds__(256) void convert_x_f16(
    const float* __restrict__ x, _Float16* __restrict__ xb)
{
  size_t t = (size_t)blockIdx.x * 256 + threadIdx.x;   // 4,194,304 threads
  const v4f* xp = (const v4f*)x + 2 * t;
  v4f a = xp[0], b = xp[1];
  v8h o;
  o[0] = (_Float16)a.x; o[1] = (_Float16)a.y; o[2] = (_Float16)a.z; o[3] = (_Float16)a.w;
  o[4] = (_Float16)b.x; o[5] = (_Float16)b.y; o[6] = (_Float16)b.z; o[7] = (_Float16)b.w;
  ((v8h*)xb)[t] = o;
}

// ------- Pass 1b: dequantize W -> f16, fold oweight into last 128 cols ------
__global__ __launch_bounds__(256) void dequant_w_f16(
    const int*   __restrict__ q,       // [N,K] values 0..15
    const float* __restrict__ scales,  // [K/128, N]
    const float* __restrict__ zeros,   // [K/128, N]
    const float* __restrict__ ow,      // [N, 128]
    _Float16*    __restrict__ wb)      // [N, K] f16 out
{
  int t  = blockIdx.x * 256 + threadIdx.x;   // N*K/8 = 2,097,152 threads
  int n  = t >> 9;                            // K/8 = 512 chunks per row
  int kc = (t & 511) << 3;                    // 8-aligned, stays in one group
  float s = scales[(size_t)(kc >> 7) * N_DIM + n];
  float z = zeros [(size_t)(kc >> 7) * N_DIM + n];
  const int4* qp = (const int4*)(q + (size_t)n * K_DIM + kc);
  int4 q0 = qp[0], q1 = qp[1];
  float w[8];
  w[0] = q0.x * s + z; w[1] = q0.y * s + z; w[2] = q0.z * s + z; w[3] = q0.w * s + z;
  w[4] = q1.x * s + z; w[5] = q1.y * s + z; w[6] = q1.z * s + z; w[7] = q1.w * s + z;
  if (kc >= K_DIM - NOUT) {                   // K-128 = 3968, 8-aligned
    const float* o = ow + (size_t)n * NOUT + (kc - (K_DIM - NOUT));
    #pragma unroll
    for (int i = 0; i < 8; ++i) w[i] += o[i];
  }
  v8h out;
  #pragma unroll
  for (int i = 0; i < 8; ++i) out[i] = (_Float16)w[i];
  *(v8h*)(wb + (size_t)n * K_DIM + kc) = out;
}

// ----------------------------- Pass 2: GEMM ---------------------------------
__global__ __launch_bounds__(256) void gemm_wmma_f16(
    const _Float16* __restrict__ A,   // xb [M,K]
    const _Float16* __restrict__ Bw,  // wb [N,K]
    const float*    __restrict__ bias,
    float*          __restrict__ C)   // [M,N]
{
  __shared__ alignas(16) _Float16 As[2][BM * LDK];
  __shared__ alignas(16) _Float16 Bs[2][BN * LDK];

  const int tid  = threadIdx.x;
  const int m0   = blockIdx.y * BM;
  const int n0   = blockIdx.x * BN;
  const int lane = tid & 31;
  const int wave = tid >> 5;
  const int wm   = wave & 3;     // 4 waves along M  -> 32 rows each
  const int wn   = wave >> 2;    // 2 waves along N  -> 64 cols each
  const int lrow = lane & 15;
  const int lhi  = lane >> 4;

  // global->LDS async staging: 256 threads, 16B chunks; 8 chunks/row, 32 rows
  const int gcol = (tid & 7) * 8;  // f16 offset within BK
  const int grow = tid >> 3;       // 0..31

  // Per-lane 32-bit global byte offsets (GVS mode: saddr base + vaddr off).
  // Advance by BK*2 = 128 bytes per K-stage.
  unsigned gAoff[4], gBoff[4];
  #pragma unroll
  for (int r = 0; r < 4; ++r) {
    gAoff[r] = (unsigned)(((m0 + grow + 32 * r) * K_DIM + gcol) * 2);
    gBoff[r] = (unsigned)(((n0 + grow + 32 * r) * K_DIM + gcol) * 2);
  }
  // LDS byte addresses for both buffers (generic-pointer low 32 bits ==
  // workgroup LDS offset per the flat-aperture mapping).
  unsigned ldsA[2][4], ldsB[2][4];
  #pragma unroll
  for (int b = 0; b < 2; ++b)
    #pragma unroll
    for (int r = 0; r < 4; ++r) {
      ldsA[b][r] = (unsigned)(size_t)&As[b][(grow + 32 * r) * LDK + gcol];
      ldsB[b][r] = (unsigned)(size_t)&Bs[b][(grow + 32 * r) * LDK + gcol];
    }

  v8f acc[2][4];
  #pragma unroll
  for (int i = 0; i < 2; ++i)
    #pragma unroll
    for (int j = 0; j < 4; ++j)
      #pragma unroll
      for (int r = 0; r < 8; ++r) acc[i][j][r] = 0.0f;

#define ASYNC_TILE(buf)                                                        \
  do {                                                                         \
    _Pragma("unroll")                                                          \
    for (int r = 0; r < 4; ++r) {                                              \
      asm volatile("global_load_async_to_lds_b128 %0, %1, %2"                  \
                   :: "v"(ldsA[buf][r]), "v"(gAoff[r]), "s"(A) : "memory");    \
      gAoff[r] += BK * 2;                                                      \
    }                                                                          \
    _Pragma("unroll")                                                          \
    for (int r = 0; r < 4; ++r) {                                              \
      asm volatile("global_load_async_to_lds_b128 %0, %1, %2"                  \
                   :: "v"(ldsB[buf][r]), "v"(gBoff[r]), "s"(Bw) : "memory");   \
      gBoff[r] += BK * 2;                                                      \
    }                                                                          \
  } while (0)

  // prologue: stage 0
  ASYNC_TILE(0);
  asm volatile("s_wait_asynccnt 0" ::: "memory");
  __syncthreads();

  const int NK = K_DIM / BK;   // 64 stages
  int cur = 0;
  for (int kt = 0; kt < NK; ++kt) {
    const bool pref = (kt + 1 < NK);
    if (pref) {
      const int nxt = cur ^ 1;
      ASYNC_TILE(nxt);     // cache -> LDS direct, overlapped with WMMAs below
    }

    // compute: two 16x16x32 K-steps over this stage
    #pragma unroll
    for (int kk = 0; kk < BK; kk += 32) {
      v16h af[2], bf[4];
      // A frag (16x32 f16): lanes 0-15 hold K [kk..kk+7] & [kk+16..kk+23],
      // lanes 16-31 hold K [kk+8..kk+15] & [kk+24..kk+31]
      #pragma unroll
      for (int i = 0; i < 2; ++i) {
        const _Float16* base =
            &As[cur][(wm * 32 + i * 16 + lrow) * LDK + kk + lhi * 8];
        v8h lo = *(const v8h*)base;
        v8h hi = *(const v8h*)(base + 16);
        #pragma unroll
        for (int e = 0; e < 8; ++e) { af[i][e] = lo[e]; af[i][e + 8] = hi[e]; }
      }
      // B frag (32x16): lane col = lrow, K [kk+lhi*8..+7] & [kk+16+lhi*8..+7]
      #pragma unroll
      for (int j = 0; j < 4; ++j) {
        const _Float16* base =
            &Bs[cur][(wn * 64 + j * 16 + lrow) * LDK + kk + lhi * 8];
        v8h lo = *(const v8h*)base;
        v8h hi = *(const v8h*)(base + 16);
        #pragma unroll
        for (int e = 0; e < 8; ++e) { bf[j][e] = lo[e]; bf[j][e + 8] = hi[e]; }
      }
      #pragma unroll
      for (int i = 0; i < 2; ++i)
        #pragma unroll
        for (int j = 0; j < 4; ++j)
          acc[i][j] = __builtin_amdgcn_wmma_f32_16x16x32_f16(
              false, af[i], false, bf[j], (short)0, acc[i][j], false, false);
    }

    if (pref) {
      // wait only for our own async loads; barrier publishes the full tile
      asm volatile("s_wait_asynccnt 0" ::: "memory");
    }
    __syncthreads();
    cur ^= 1;
  }
#undef ASYNC_TILE

  // epilogue: C/D layout -> lane holds col = lrow, rows lhi*8 + r
  #pragma unroll
  for (int j = 0; j < 4; ++j) {
    const int col = n0 + wn * 64 + j * 16 + lrow;
    const float bv = bias[col];
    #pragma unroll
    for (int i = 0; i < 2; ++i) {
      const int row = m0 + wm * 32 + i * 16 + lhi * 8;
      float* cp = C + (size_t)row * N_DIM + col;
      #pragma unroll
      for (int r = 0; r < 8; ++r) cp[(size_t)r * N_DIM] = acc[i][j][r] + bv;
    }
  }
}

// ---------------------------------------------------------------------------
extern "C" void kernel_launch(void* const* d_in, const int* in_sizes, int n_in,
                              void* d_out, int out_size, void* d_ws, size_t ws_size,
                              hipStream_t stream) {
  const float* x     = (const float*)d_in[0];
  const int*   qw    = (const int*)d_in[1];
  const float* sc    = (const float*)d_in[2];
  const float* zz    = (const float*)d_in[3];
  const float* ow    = (const float*)d_in[4];
  const float* bias  = (const float*)d_in[5];
  float*       out   = (float*)d_out;

  _Float16* xb = (_Float16*)d_ws;                                      // 64 MiB
  _Float16* wb = (_Float16*)((char*)d_ws + (size_t)M_DIM * K_DIM * 2); // +32 MiB

  convert_x_f16<<<(M_DIM * (K_DIM / 8)) / 256, 256, 0, stream>>>(x, xb);
  dequant_w_f16<<<(N_DIM * (K_DIM / 8)) / 256, 256, 0, stream>>>(qw, sc, zz, ow, wb);

  dim3 grid(N_DIM / BN, M_DIM / BM);   // 32 x 64
  gemm_wmma_f16<<<grid, 256, 0, stream>>>(xb, wb, bias, out);
}
